// SDTs_80298708566248
// MI455X (gfx1250) — matrix operand order repaired and checked by reading, not compiled, not run on hardware
//
#include <hip/hip_runtime.h>

typedef __attribute__((ext_vector_type(16))) __bf16 v16bf;
typedef __attribute__((ext_vector_type(8)))  float  v8f;
typedef __attribute__((ext_vector_type(4)))  float  v4f;
typedef __attribute__((ext_vector_type(4)))  int    v4i;

// Problem constants (match reference)
constexpr int T_TREES = 100;
constexpr int PAIRS   = T_TREES / 2;   // 50 tree pairs, 2 trees per 16-col WMMA tile
constexpr int NX      = 256;           // K dimension
constexpr int KTILES  = NX / 32;       // 8 WMMA K-steps of 32
constexpr int NROWS   = 32768;         // N samples (M dimension)
constexpr int INNER   = 7;
constexpr int LEAFS   = 8;
constexpr int WPB     = 8;             // waves per block
constexpr int PAIR_ELEMS = KTILES * 32 * 16;  // bf16 elems per packed tree-pair (4096 = 8KB)

// --- gfx1250 async global->LDS path (guarded; falls back to sync copy) -----
// Builtin signature (from hipcc diagnostic): (v4i AS1* src, v4i AS3* dst, Ii offset, Ii cpol)
#define GAS1_V4I(p) ((__attribute__((address_space(1))) v4i*)(p))
#define LAS3_V4I(p) ((__attribute__((address_space(3))) v4i*)(p))

#if __has_builtin(__builtin_amdgcn_global_load_async_to_lds_b128)
#define HAVE_ASYNC_LDS 1
#else
#define HAVE_ASYNC_LDS 0
#endif

__device__ __forceinline__ void async_wait0() {
#if __has_builtin(__builtin_amdgcn_s_wait_asynccnt)
    __builtin_amdgcn_s_wait_asynccnt(0);
#elif HAVE_ASYNC_LDS
    asm volatile("s_wait_asynccnt 0x0" ::: "memory");
#endif
}

// Copy the 8KB packed-B block for tree-pair `pr` into the LDS stage buffer;
// 256 threads x 32B each (two async B128 transfers per thread).
__device__ __forceinline__ void stage_pair(const __bf16* __restrict__ wp,
                                           __bf16* __restrict__ dst_base,
                                           int pr, int tid) {
    const char* gsrc = (const char*)(wp + (size_t)pr * PAIR_ELEMS) + (size_t)tid * 32;
    char*       ldst = (char*)(dst_base + tid * 16);
#if HAVE_ASYNC_LDS
    __builtin_amdgcn_global_load_async_to_lds_b128(GAS1_V4I(gsrc), LAS3_V4I(ldst), 0,  0);
    __builtin_amdgcn_global_load_async_to_lds_b128(GAS1_V4I(gsrc), LAS3_V4I(ldst), 16, 0);
#else
    *(v16bf*)ldst = *(const v16bf*)gsrc;
#endif
}

// ---------------------------------------------------------------------------
// Kernel 1: mean(Y) -> ws[0]
// ---------------------------------------------------------------------------
__global__ void ymean_kernel(const float* __restrict__ Y, float* __restrict__ m, int n) {
    __shared__ float s[256];
    float a = 0.f;
    for (int i = threadIdx.x; i < n; i += 256) a += Y[i];
    s[threadIdx.x] = a;
    __syncthreads();
    for (int w = 128; w > 0; w >>= 1) {
        if ((int)threadIdx.x < w) s[threadIdx.x] += s[threadIdx.x + w];
        __syncthreads();
    }
    if (threadIdx.x == 0) m[0] = s[0] / (float)n;
}

// ---------------------------------------------------------------------------
// Kernel 2: pack W (T,256,7) f32 -> bf16 in WMMA B-fragment order.
//   B 32x16 bf16 fragment: h=lane>>4; e<8 -> K = 8h+e, e>=8 -> K = 16+8h+(e-8); col = lane&15
//   col 0..6  -> tree 2*pair;  col 8..14 -> tree 2*pair+1;  col 7,15 -> zero
// ---------------------------------------------------------------------------
__global__ void pack_w_kernel(const float* __restrict__ W, __bf16* __restrict__ wp) {
    int idx  = blockIdx.x * 256 + threadIdx.x;   // < PAIRS * 4096
    int e    = idx & 15;
    int lane = (idx >> 4) & 31;
    int kt   = (idx >> 9) & 7;
    int pr   = idx >> 12;
    int col  = lane & 15;
    int kloc = ((e >> 3) << 4) + ((lane >> 4) << 3) + (e & 7);
    int k    = kt * 32 + kloc;
    float v = 0.f;
    if (col < INNER)                    v = W[((2 * pr    ) * NX + k) * INNER + col];
    else if (col >= 8 && col < 8+INNER) v = W[((2 * pr + 1) * NX + k) * INNER + (col - 8)];
    wp[idx] = (__bf16)v;
}

// ---------------------------------------------------------------------------
// Kernel 3: fused GEMM (bf16 WMMA) + sigmoid + tree leaf products + phi dot.
// Each wave owns one 16-row M-tile; A fragments stay in VGPRs across all 50
// tree pairs (X read exactly once). B blocks are double-buffered in LDS via
// async global->LDS copies shared by the 8 waves of the block.
// ---------------------------------------------------------------------------
__global__ void __launch_bounds__(256) sdt_fused_kernel(
    const float*  __restrict__ X,
    const __bf16* __restrict__ wp,
    const float*  __restrict__ b,
    const float*  __restrict__ phi,
    const float*  __restrict__ meanp,
    float*        __restrict__ out)
{
    __shared__ __align__(32) __bf16 bstage[2][PAIR_ELEMS];   // 16 KB double buffer
    __shared__ float tpose[WPB][16][16];                     //  8 KB transpose
    __shared__ float s_b[T_TREES * INNER];                   // 2.8 KB
    __shared__ float s_phi[T_TREES * LEAFS];                 // 3.2 KB

    const int tid   = threadIdx.x;
    const int lane  = tid & 31;
    const int wv    = tid >> 5;
    const int l16   = lane & 15;
    const int half  = lane >> 4;                     // 0 or 1
    const int mtile = blockIdx.x * WPB + wv;
    const long rowbase = (long)(mtile * 16 + l16) * NX;
    const int  kbase   = half * 8;                   // A-frag K base within 16-group

    // Preload b / phi into LDS (once per block)
    for (int i = tid; i < T_TREES * INNER; i += 256) s_b[i]   = b[i];
    for (int i = tid; i < T_TREES * LEAFS; i += 256) s_phi[i] = phi[i];

    // Stage tree-pair 0's B block
    stage_pair(wp, &bstage[0][0], 0, tid);

    // --- Load & convert A fragments (16-bit A 16x32 VGPR layout) ---
    // element e<8 -> K = kbase + e ; element e>=8 -> K = 16 + kbase + (e-8)
    v16bf afrag[KTILES];
    #pragma unroll
    for (int kt = 0; kt < KTILES; ++kt) {
        const float* p = X + rowbase + kt * 32 + kbase;
        v4f f0 = *(const v4f*)(p);
        v4f f1 = *(const v4f*)(p + 4);
        v4f f2 = *(const v4f*)(p + 16);
        v4f f3 = *(const v4f*)(p + 20);
        v16bf a;
        #pragma unroll
        for (int j = 0; j < 4; ++j) {
            a[j]      = (__bf16)f0[j];
            a[4 + j]  = (__bf16)f1[j];
            a[8 + j]  = (__bf16)f2[j];
            a[12 + j] = (__bf16)f3[j];
        }
        afrag[kt] = a;
    }

    const float mean = *meanp;
    float acc = 0.f;

    async_wait0();
    __syncthreads();   // pair-0 stage + b/phi visible to all waves

    for (int pr = 0; pr < PAIRS; ++pr) {
        const int cur = pr & 1;
        // Kick off async copy of the next pair's B block into the other buffer
        if (pr + 1 < PAIRS) stage_pair(wp, &bstage[cur ^ 1][0], pr + 1, tid);

        // Preload all 8 B fragments into distinct VGPRs so the ds_load latency
        // overlaps the serial WMMA chain (looser dscnt waits instead of 0).
        v16bf bfrag[KTILES];
        #pragma unroll
        for (int kt = 0; kt < KTILES; ++kt)
            bfrag[kt] = *(const v16bf*)&bstage[cur][(kt * 32 + lane) * 16];

        // 8 chained WMMA steps over K=256
        v8f c = {};
        #pragma unroll
        for (int kt = 0; kt < KTILES; ++kt)
            c = __builtin_amdgcn_wmma_f32_16x16x32_bf16(
                    /*neg_a=*/false, afrag[kt], /*neg_b=*/false, bfrag[kt],
                    /*c_mod=*/(short)0, c, /*reuse_a=*/false, /*reuse_b=*/false);

        // Transpose 16x16 C through this wave's private LDS slice.
        // Cross-lane traffic stays within the wave: hardware keeps same-wave
        // LDS ops in order, so a compiler scheduling barrier suffices
        // (no block-wide s_barrier needed here).
        #pragma unroll
        for (int j = 0; j < 8; ++j) tpose[wv][j + half * 8][l16] = c[j];
        __builtin_amdgcn_wave_barrier();

        // Each lane evaluates one (row=l16, tree=2*pr+half)
        const int t = 2 * pr + half;
        float sg[INNER];
        #pragma unroll
        for (int j = 0; j < INNER; ++j) {
            float x = tpose[wv][l16][half * 8 + j] + s_b[t * INNER + j];
            sg[j] = 1.f / (1.f + __expf(-x));
        }
        __builtin_amdgcn_wave_barrier();   // keep next-iter stores after reads
        float o = 0.f;
        #pragma unroll
        for (int i = 0; i < LEAFS; ++i) {
            float f1 = (i & 1)        ? sg[3 + (i >> 1)] : 1.f - sg[3 + (i >> 1)];
            float f2 = ((i >> 1) & 1) ? sg[1 + (i >> 2)] : 1.f - sg[1 + (i >> 2)];
            float f3 = ((i >> 2) & 1) ? sg[0]            : 1.f - sg[0];
            o += s_phi[t * LEAFS + i] * f1 * f2 * f3;
        }
        acc += o;

        async_wait0();     // own wave's async copies for next buffer done
        __syncthreads();   // publish staged B buffer block-wide; bstage WAR safe
    }

    // Fold the two tree-slots (lane halves) and write 16 rows per wave
    float other = __shfl_xor(acc, 16, 32);
    if (half == 0) out[mtile * 16 + l16] = mean + 0.1f * (acc + other);
}

// ---------------------------------------------------------------------------
extern "C" void kernel_launch(void* const* d_in, const int* in_sizes, int n_in,
                              void* d_out, int out_size, void* d_ws, size_t ws_size,
                              hipStream_t stream) {
    const float* X   = (const float*)d_in[0];
    const float* Y   = (const float*)d_in[1];
    const float* W   = (const float*)d_in[2];
    const float* b   = (const float*)d_in[3];
    const float* phi = (const float*)d_in[4];
    float* out = (float*)d_out;

    float*  meanp = (float*)d_ws;                       // 1 float (padded to 256B)
    __bf16* wpack = (__bf16*)((char*)d_ws + 256);       // 50*4096 bf16 = 400 KB

    // 1) mean(Y)
    ymean_kernel<<<1, 256, 0, stream>>>(Y, meanp, NROWS);

    // 2) pack W into WMMA-B order (PAIRS*4096 elems / 256 = 800 blocks)
    pack_w_kernel<<<(PAIRS * PAIR_ELEMS) / 256, 256, 0, stream>>>(W, wpack);

    // 3) fused GEMM + tree evaluation: 2048 M-tiles / 8 waves = 256 blocks
    sdt_fused_kernel<<<(NROWS / 16) / WPB, 256, 0, stream>>>(X, wpack, b, phi, meanp, out);
}
